// lstm_5652176962240
// MI455X (gfx1250) — matrix operand order repaired and checked by reading, not compiled, not run on hardware
//
#include <hip/hip_runtime.h>
#include <hip/hip_bf16.h>

// ---------------------------------------------------------------------------
// LSTM  B=128, T=2048, I=128, H=128  (gates 4H=512)
//   Pass A: convert W_ih/W_hh -> bf16, fuse biases (b_ih+b_hh)
//   Pass B: xg = xs @ W_ih^T + bias; 32-t-wide blocks, coalesced float4
//           staging into 128KB LDS, W_ih fragments reused across 32 tiles.
//   Pass C: 8 persistent blocks (one 16-row batch tile each) run the 2048-step
//           recurrence with v_wmma_f32_16x16x32_bf16; W_hh fragments pinned in
//           VGPRs, h in LDS, c in VGPRs, xg software-pipelined + prefetched.
//   Pass D: tiny head  relu(h@W_out^T+b_out)@W_cls^T + b_cls -> [B,1]
// ---------------------------------------------------------------------------

typedef __attribute__((ext_vector_type(16))) __bf16 v16bf;
typedef __attribute__((ext_vector_type(8)))  __bf16 v8bf;
typedef __attribute__((ext_vector_type(8)))  float  v8f;
typedef __attribute__((ext_vector_type(4)))  float  v4f;

#define LSTM_B 128
#define LSTM_T 2048
#define LSTM_I 128
#define LSTM_H 128
#define LSTM_G 512  // 4H
#define TT     32   // t-tile width in pass B

// xg tiled layout: element (t, bTile, gateCol, m) ; m = batch row within tile
__device__ __host__ inline size_t xgoff(int t, int bt, int g, int m) {
    return ((((size_t)t * 8 + bt) * LSTM_G) + g) * 16 + m;
}

// ---- WMMA helpers ---------------------------------------------------------

__device__ inline v8f wmma_bf16(v16bf a, v16bf b, v8f c) {
    return __builtin_amdgcn_wmma_f32_16x16x32_bf16(
        false, a, false, b, (short)0, c, false, false);
}

// A-fragment (16x32 bf16) gathered from a [16][128] (row m, col k) bf16 array.
// Lane L: m=L&15, K = kc*32 + (L>>4)*16 .. +15 (contiguous, matches B).
__device__ inline v16bf load_a_frag(const __bf16* sh, int lane, int kc) {
    int m  = lane & 15;
    int hi = lane >> 4;
    const v8bf lo = *(const v8bf*)&sh[m * 128 + kc * 32 + hi * 16];
    const v8bf hh = *(const v8bf*)&sh[m * 128 + kc * 32 + hi * 16 + 8];
    return __builtin_shufflevector(lo, hh, 0, 1, 2, 3, 4, 5, 6, 7,
                                           8, 9, 10, 11, 12, 13, 14, 15);
}

// B-fragment (32x16 bf16) from row-major W[g][k] (k contiguous, 128 per row).
// Lane L: n = L&15 (column), K = kc*32 + (L>>4)*16 .. +15  -> one 32B load.
__device__ inline v16bf load_b_frag(const __bf16* __restrict__ w, int row,
                                    int lane, int kc) {
    int kb = (lane >> 4) * 16;
    return *(const v16bf*)&w[(size_t)row * 128 + kc * 32 + kb];
}

__device__ inline float sigmoid_f(float v) {
    return 1.0f / (1.0f + __expf(-v));
}
__device__ inline float tanh_f(float v) {
    v = fminf(fmaxf(v, -15.0f), 15.0f);
    float e = __expf(2.0f * v);
    return (e - 1.0f) / (e + 1.0f);
}

// ---- Pass A: weight conversion -------------------------------------------

__global__ void lstm_prep_kernel(const float* __restrict__ W_ih,
                                 const float* __restrict__ W_hh,
                                 const float* __restrict__ b_ih,
                                 const float* __restrict__ b_hh,
                                 __bf16* __restrict__ wih_bf,
                                 __bf16* __restrict__ whh_bf,
                                 float* __restrict__ biasC) {
    int idx = blockIdx.x * 256 + threadIdx.x;
    if (idx < LSTM_G * LSTM_I) {
        wih_bf[idx] = (__bf16)W_ih[idx];
        whh_bf[idx] = (__bf16)W_hh[idx];
    }
    if (idx < LSTM_G) biasC[idx] = b_ih[idx] + b_hh[idx];
}

// ---- Pass B: xg = xs @ W_ih^T + bias  ------------------------------------
// grid (T/32, B/16), block 256 (8 waves). Block stages x[b-tile, :, t0..t0+31]
// with coalesced float4 loads into LDS [t][m][k] bf16, then each wave emits
// 32 t-tiles x 4 gate-tiles reusing its 16 pinned W_ih fragments.

__global__ void __launch_bounds__(256, 1)
lstm_xgate_kernel(const float* __restrict__ x,
                  const __bf16* __restrict__ wih_bf,
                  const float* __restrict__ biasC,
                  __bf16* __restrict__ xg) {
    __shared__ __bf16 xa[TT * 16 * 128];  // 128KB: [tt][m][k]

    const int t0 = blockIdx.x * TT;
    const int bt = blockIdx.y;
    const int b0 = bt * 16;
    const int tid = threadIdx.x;

    // Stage: x layout [B,1,I,T] -> x[b*I*T + k*T + t]; read float4 along t.
    for (int v = tid; v < 16 * 128 * (TT / 4); v += 256) {
        int tt4 = v & 7;           // which float4 along t (TT/4 = 8)
        int k   = (v >> 3) & 127;
        int m   = v >> 10;
        v4f val = *(const v4f*)&x[((size_t)(b0 + m) * LSTM_I + k) * LSTM_T +
                                  t0 + tt4 * 4];
#pragma unroll
        for (int j = 0; j < 4; ++j)
            xa[(tt4 * 4 + j) * 2048 + m * 128 + k] = (__bf16)val[j];
    }
    __syncthreads();

    const int lane = tid & 31;
    const int w    = tid >> 5;
    const int nn   = lane & 15;
    const int hi   = lane >> 4;

    // loop-invariant W_ih fragments + biases (pinned across 32 t-tiles)
    v16bf bfr[4][4];
    float bv[4];
#pragma unroll
    for (int nt = 0; nt < 4; ++nt) {
        int g0 = w * 64 + nt * 16;
        bv[nt] = biasC[g0 + nn];
#pragma unroll
        for (int kc = 0; kc < 4; ++kc)
            bfr[nt][kc] = load_b_frag(wih_bf, g0 + nn, lane, kc);
    }

#pragma unroll 1
    for (int tt = 0; tt < TT; ++tt) {
        v16bf a[4];
#pragma unroll
        for (int kc = 0; kc < 4; ++kc)
            a[kc] = load_a_frag(xa + tt * 2048, lane, kc);

#pragma unroll
        for (int nt = 0; nt < 4; ++nt) {
            int g0 = w * 64 + nt * 16;
            v8f acc = {};
#pragma unroll
            for (int kc = 0; kc < 4; ++kc)
                acc = wmma_bf16(a[kc], bfr[nt][kc], acc);
            v8bf pk;
#pragma unroll
            for (int r = 0; r < 8; ++r) pk[r] = (__bf16)(acc[r] + bv[nt]);
            // lane owns (m = r + 8*hi, n = nn): contiguous 16B store
            *(v8bf*)&xg[xgoff(t0 + tt, bt, g0 + nn, 8 * hi)] = pk;
        }
    }
}

// ---- Pass C: recurrence ---------------------------------------------------
// grid 8 (batch tiles), block 256 (8 waves). Wave w owns hidden columns
// w*16..w*16+15 and computes the i/f/g/o gate tiles for those columns.

__global__ void __launch_bounds__(256, 1)
__attribute__((amdgpu_waves_per_eu(1, 2)))
lstm_recur_kernel(const float* __restrict__ h0,
                  const float* __restrict__ c0,
                  const __bf16* __restrict__ whh_bf,
                  const __bf16* __restrict__ xg,
                  float* __restrict__ hT) {
    __shared__ __bf16 hl[16 * 128];  // h state, [m][k] bf16

    const int bt  = blockIdx.x;
    const int b0  = bt * 16;
    const int tid = threadIdx.x;
    const int lane = tid & 31;
    const int w    = tid >> 5;
    const int nn   = lane & 15;
    const int hi   = lane >> 4;
    const int col  = w * 16 + nn;  // hidden column owned by this lane

    // init h state in LDS from h0 [1,B,H]
    for (int idx = tid; idx < 16 * 128; idx += 256) {
        int m = idx >> 7, k = idx & 127;
        hl[idx] = (__bf16)h0[(size_t)(b0 + m) * LSTM_H + k];
    }

    // c state in registers: element r -> (m = r + 8*hi, n = col)
    float cst[8];
#pragma unroll
    for (int r = 0; r < 8; ++r)
        cst[r] = c0[(size_t)(b0 + r + 8 * hi) * LSTM_H + col];

    // W_hh fragments pinned in VGPRs: gate gi in {i,f,g,o}, rows gi*128+col
    v16bf bfr[4][4];
#pragma unroll
    for (int gi = 0; gi < 4; ++gi)
#pragma unroll
        for (int kc = 0; kc < 4; ++kc)
            bfr[gi][kc] = load_b_frag(whh_bf, gi * 128 + col, lane, kc);

    // software-pipelined xg tile loads (one step ahead)
    v8bf xcur[4];
#pragma unroll
    for (int gi = 0; gi < 4; ++gi)
        xcur[gi] = *(const v8bf*)&xg[xgoff(0, bt, gi * 128 + col, 8 * hi)];

    float hout[8];
    __syncthreads();

#pragma unroll 1
    for (int t = 0; t < LSTM_T; ++t) {
        // prefetch far ahead into caches
        if (t + 16 < LSTM_T) {
#pragma unroll
            for (int gi = 0; gi < 4; ++gi)
                __builtin_prefetch(
                    &xg[xgoff(t + 16, bt, gi * 128 + col, 8 * hi)], 0, 0);
        }
        // issue next-step loads (consumed at end of iteration)
        v8bf xnxt[4];
        int tn = (t + 1 < LSTM_T) ? t + 1 : t;
#pragma unroll
        for (int gi = 0; gi < 4; ++gi)
            xnxt[gi] = *(const v8bf*)&xg[xgoff(tn, bt, gi * 128 + col, 8 * hi)];

        // A fragments from current h
        v16bf a[4];
#pragma unroll
        for (int kc = 0; kc < 4; ++kc) a[kc] = load_a_frag(hl, lane, kc);

        // gates = xg_t + h @ W_hh^T   (bias already folded into xg)
        v8f acc[4];
#pragma unroll
        for (int gi = 0; gi < 4; ++gi) {
            v8f c = {};
#pragma unroll
            for (int r = 0; r < 8; ++r) c[r] = (float)xcur[gi][r];
#pragma unroll
            for (int kc = 0; kc < 4; ++kc)
                c = wmma_bf16(a[kc], bfr[gi][kc], c);
            acc[gi] = c;
        }

        // elementwise cell update (PyTorch order i,f,g,o)
#pragma unroll
        for (int r = 0; r < 8; ++r) {
            float ig = sigmoid_f(acc[0][r]);
            float fg = sigmoid_f(acc[1][r]);
            float gg = tanh_f(acc[2][r]);
            float og = sigmoid_f(acc[3][r]);
            cst[r]  = fg * cst[r] + ig * gg;
            hout[r] = og * tanh_f(cst[r]);
        }

        __syncthreads();  // everyone done reading hl
#pragma unroll
        for (int r = 0; r < 8; ++r)
            hl[(r + 8 * hi) * 128 + col] = (__bf16)hout[r];
        __syncthreads();  // hl updated for next step

#pragma unroll
        for (int gi = 0; gi < 4; ++gi) xcur[gi] = xnxt[gi];
    }

    // write final hidden state (fp32) for the head
#pragma unroll
    for (int r = 0; r < 8; ++r)
        hT[(size_t)(b0 + r + 8 * hi) * LSTM_H + col] = hout[r];
}

// ---- Pass D: classification head -----------------------------------------

__global__ void lstm_head_kernel(const float* __restrict__ hT,
                                 const float* __restrict__ W_out,
                                 const float* __restrict__ b_out,
                                 const float* __restrict__ W_cls,
                                 const float* __restrict__ b_cls,
                                 float* __restrict__ out) {
    int b = threadIdx.x;  // 128 threads
    float hrow[LSTM_H];
#pragma unroll 8
    for (int k = 0; k < LSTM_H; ++k) hrow[k] = hT[(size_t)b * LSTM_H + k];
    float y = 0.0f;
    for (int j = 0; j < LSTM_H; ++j) {
        float s = b_out[j];
#pragma unroll 8
        for (int k = 0; k < LSTM_H; ++k)
            s += hrow[k] * W_out[(size_t)j * LSTM_H + k];
        y += fmaxf(s, 0.0f) * W_cls[j];
    }
    out[b] = y + b_cls[0];
}

// ---- Launch ---------------------------------------------------------------

extern "C" void kernel_launch(void* const* d_in, const int* in_sizes, int n_in,
                              void* d_out, int out_size, void* d_ws,
                              size_t ws_size, hipStream_t stream) {
    const float* x     = (const float*)d_in[0];
    const float* h0    = (const float*)d_in[1];
    const float* c0    = (const float*)d_in[2];
    const float* W_ih  = (const float*)d_in[3];
    const float* W_hh  = (const float*)d_in[4];
    const float* b_ih  = (const float*)d_in[5];
    const float* b_hh  = (const float*)d_in[6];
    const float* W_out = (const float*)d_in[7];
    const float* b_out = (const float*)d_in[8];
    const float* W_cls = (const float*)d_in[9];
    const float* b_cls = (const float*)d_in[10];
    float* out = (float*)d_out;

    // workspace layout
    const size_t XG_ELEMS = (size_t)LSTM_T * 8 * LSTM_G * 16;  // 134,217,728
    char* ws = (char*)d_ws;
    __bf16* xg     = (__bf16*)ws;
    __bf16* wih_bf = (__bf16*)(ws + XG_ELEMS * 2);
    __bf16* whh_bf = wih_bf + (size_t)LSTM_G * LSTM_I;
    float*  biasC  = (float*)(whh_bf + (size_t)LSTM_G * LSTM_H);
    float*  hT     = biasC + LSTM_G;

    lstm_prep_kernel<<<dim3(256), dim3(256), 0, stream>>>(
        W_ih, W_hh, b_ih, b_hh, wih_bf, whh_bf, biasC);

    lstm_xgate_kernel<<<dim3(LSTM_T / TT, LSTM_B / 16), dim3(256), 0, stream>>>(
        x, wih_bf, biasC, xg);

    lstm_recur_kernel<<<dim3(LSTM_B / 16), dim3(256), 0, stream>>>(
        h0, c0, whh_bf, xg, hT);

    lstm_head_kernel<<<dim3(1), dim3(128), 0, stream>>>(
        hT, W_out, b_out, W_cls, b_cls, out);
}
